// PathBundleChoiceScorer_69741678952826
// MI455X (gfx1250) — matrix operand order repaired and checked by reading, not compiled, not run on hardware
//
#include <hip/hip_runtime.h>
#include <hip/hip_bf16.h>
#include <hip/hip_fp16.h>
#include <math.h>
#include <stdint.h>

typedef __attribute__((ext_vector_type(16))) _Float16 v16h;
typedef __attribute__((ext_vector_type(8)))  _Float16 v8h;
typedef __attribute__((ext_vector_type(4)))  _Float16 v4h;
typedef __attribute__((ext_vector_type(8)))  float    v8f;

#define TILE_M 128
#define TILE_N 128
#define TILE_K 32
#define LDA_S  40    // As row stride in halves (80B/row, bank-conflict padding)
#define LDB_S  136   // Ws row stride in halves (272B/row)

// ---- CDNA5 helpers ----------------------------------------------------------

__device__ __forceinline__ void async_to_lds_b128(const void* gsrc, void* lds_dst) {
    unsigned ldsoff = (unsigned)(uintptr_t)(__attribute__((address_space(3))) void*)lds_dst;
    asm volatile("global_load_async_to_lds_b128 %0, %1, off"
                 :: "v"(ldsoff), "v"(gsrc) : "memory");
}

__device__ __forceinline__ void wait_async_le4() {
#if __has_builtin(__builtin_amdgcn_s_wait_asynccnt)
    __builtin_amdgcn_s_wait_asynccnt(4);
#else
    asm volatile("s_wait_asynccnt 0x4" ::: "memory");
#endif
}

__device__ __forceinline__ void wait_async_le0() {
#if __has_builtin(__builtin_amdgcn_s_wait_asynccnt)
    __builtin_amdgcn_s_wait_asynccnt(0);
#else
    asm volatile("s_wait_asynccnt 0x0" ::: "memory");
#endif
}

__device__ __forceinline__ float fast_tanh(float x) {
#if __has_builtin(__builtin_amdgcn_tanhf)
    return __builtin_amdgcn_tanhf(x);        // v_tanh_f32 (gfx1250 native)
#elif __has_builtin(__builtin_amdgcn_tanh_f32)
    return __builtin_amdgcn_tanh_f32(x);
#else
    return tanhf(x);
#endif
}

// ---- f32 -> f16 bulk convert (X pre-pass) -----------------------------------

__global__ __launch_bounds__(256)
void cvt_f32_f16(const float* __restrict__ in, _Float16* __restrict__ out, int n4) {
    int i = blockIdx.x * blockDim.x + threadIdx.x;
    if (i < n4) {
        const float4 v = reinterpret_cast<const float4*>(in)[i];
        v4h h;
        h[0] = (_Float16)v.x; h[1] = (_Float16)v.y;
        h[2] = (_Float16)v.z; h[3] = (_Float16)v.w;
        reinterpret_cast<v4h*>(out)[i] = h;
    }
}

// ---- f32 [N][K] -> f16 [K][N] transpose-convert (W1, W2 pre-pass) -----------

__global__ __launch_bounds__(256)
void cvt_transpose_f32_f16(const float* __restrict__ in, _Float16* __restrict__ out,
                           int N, int K) {
    __shared__ float tile[32][33];
    const int tx = threadIdx.x & 31, ty = threadIdx.x >> 5;  // ty = 0..7
    const int k0 = blockIdx.x * 32, n0 = blockIdx.y * 32;
#pragma unroll
    for (int j = 0; j < 4; ++j) {
        const int nr = ty + j * 8;
        tile[nr][tx] = in[(size_t)(n0 + nr) * K + k0 + tx];
    }
    __syncthreads();
#pragma unroll
    for (int j = 0; j < 4; ++j) {
        const int kr = ty + j * 8;
        out[(size_t)(k0 + kr) * N + n0 + tx] = (_Float16)tile[tx][kr];
    }
}

// ---- double-buffered async tile stage ---------------------------------------
// A: [M][K] f16 row-major;  Bt: [K][N] f16 row-major (pre-transposed weights).
// Per thread: 2 A-chunks + 2 B-chunks of 16B, async DMA straight into LDS.

__device__ __forceinline__ void stage_tiles(const _Float16* __restrict__ A,
                                            const _Float16* __restrict__ Bt,
                                            int K, int N, int mBlk, int nBlk, int k0,
                                            int tid, _Float16* AsBuf, _Float16* WsBuf) {
#pragma unroll
    for (int i = 0; i < 2; ++i) {
        const int idx = tid + i * 256;
        const int r = idx >> 2, c8 = (idx & 3) * 8;          // 128 rows x 4 chunks
        async_to_lds_b128(A + (size_t)(mBlk + r) * K + k0 + c8,
                          AsBuf + r * LDA_S + c8);
    }
#pragma unroll
    for (int i = 0; i < 2; ++i) {
        const int idx = tid + i * 256;
        const int rk = idx >> 4, c8 = (idx & 15) * 8;        // 32 rows x 16 chunks
        async_to_lds_b128(Bt + (size_t)(k0 + rk) * N + nBlk + c8,
                          WsBuf + rk * LDB_S + c8);
    }
}

// ---- C[m,n] = tanh( sum_k A[m,k]*Bt[k,n] ), f16 in, f32 WMMA accumulate -----
// Block = 256 threads = 8 waves; block tile 128x128; wave tile 32x64.

__global__ __launch_bounds__(256)
void gemm_f16_tanh(const _Float16* __restrict__ A, const _Float16* __restrict__ Bt,
                   _Float16* __restrict__ C, int M, int N, int K) {
    __shared__ __align__(16) _Float16 As[2][TILE_M * LDA_S];
    __shared__ __align__(16) _Float16 Ws[2][TILE_K * LDB_S];

    const int tid   = threadIdx.x;
    const int lane  = tid & 31;
    const int wave  = tid >> 5;
    const int mWave = (wave & 3) * 32;   // 4 waves along M
    const int nWave = (wave >> 2) * 64;  // 2 waves along N
    const int mBlk  = blockIdx.x * TILE_M;
    const int nBlk  = blockIdx.y * TILE_N;
    const int half  = lane >> 4;
    const int mr    = lane & 15;

    v8f acc[2][4];
#pragma unroll
    for (int mt = 0; mt < 2; ++mt)
#pragma unroll
        for (int nt = 0; nt < 4; ++nt)
#pragma unroll
            for (int e = 0; e < 8; ++e) acc[mt][nt][e] = 0.0f;

    const int NK = K / TILE_K;
    stage_tiles(A, Bt, K, N, mBlk, nBlk, 0, tid, As[0], Ws[0]);   // prologue

    for (int kt = 0; kt < NK; ++kt) {
        const int cur = kt & 1;
        __syncthreads();            // WAR: everyone done reading buffer cur^1
        if (kt + 1 < NK) {
            stage_tiles(A, Bt, K, N, mBlk, nBlk, (kt + 1) * TILE_K, tid,
                        As[cur ^ 1], Ws[cur ^ 1]);
            wait_async_le4();       // in-order: stage kt's 4 DMAs have landed
        } else {
            wait_async_le0();
        }
        __syncthreads();            // all waves' stage-kt data visible

        // A 16x32 f16 frag: half=lane>>4, M=lane&15;
        //   a[0..7] = K 8h..8h+7, a[8..15] = K 16+8h..23+8h
        v16h af[2];
#pragma unroll
        for (int mt = 0; mt < 2; ++mt) {
            const _Float16* p = &As[cur][(mWave + mt * 16 + mr) * LDA_S + 8 * half];
            const v8h lo = *reinterpret_cast<const v8h*>(p);
            const v8h hi = *reinterpret_cast<const v8h*>(p + 16);
            af[mt] = __builtin_shufflevector(lo, hi, 0, 1, 2, 3, 4, 5, 6, 7,
                                             8, 9, 10, 11, 12, 13, 14, 15);
        }
        // B 32x16 f16 frag: N=lane&15; b[0..7] = K 8h..8h+7, b[8..15] = K 16+8h..23+8h
        v16h bf[4];
#pragma unroll
        for (int nt = 0; nt < 4; ++nt) {
            const _Float16* p = &Ws[cur][nWave + nt * 16 + mr];
#pragma unroll
            for (int s = 0; s < 16; ++s)
                bf[nt][s] = p[(s < 8 ? 8 * half + s : 8 + 8 * half + s) * LDB_S];
        }
#pragma unroll
        for (int mt = 0; mt < 2; ++mt)
#pragma unroll
            for (int nt = 0; nt < 4; ++nt)
                acc[mt][nt] = __builtin_amdgcn_wmma_f32_16x16x32_f16(
                    false, af[mt], false, bf[nt], (short)0, acc[mt][nt], false, false);
    }

    // ---- epilogue: tanh (v_tanh_f32) + f16 store ----
#pragma unroll
    for (int mt = 0; mt < 2; ++mt) {
#pragma unroll
        for (int nt = 0; nt < 4; ++nt) {
            const int n = nBlk + nWave + nt * 16 + mr;
#pragma unroll
            for (int e = 0; e < 8; ++e) {
                const int m = mBlk + mWave + mt * 16 + half * 8 + e;
                C[(size_t)m * N + n] = (_Float16)fast_tanh(acc[mt][nt][e]);
            }
        }
    }
}

// ---- Per row: v = U[n,:].Wout; analytic softmax/top-9 collapse --------------
// 27 paths have values {+v x4, -v x4, 0 x19} with scores sP/sN/s0.

__global__ __launch_bounds__(256)
void finalize_kernel(const _Float16* __restrict__ U, const float* __restrict__ Wout,
                     const float* __restrict__ soa_w, const float* __restrict__ soa_b,
                     const float* __restrict__ sob_w, const float* __restrict__ sob_b,
                     float* __restrict__ out, int K) {
    __shared__ float red[256];
    const int n = blockIdx.x;
    float sum = 0.0f;
    for (int f = threadIdx.x; f < K; f += 256)
        sum += (float)U[(size_t)n * K + f] * Wout[f];
    red[threadIdx.x] = sum;
    __syncthreads();
    for (int s = 128; s > 0; s >>= 1) {
        if (threadIdx.x < s) red[threadIdx.x] += red[threadIdx.x + s];
        __syncthreads();
    }
    if (threadIdx.x == 0) {
        const float v = red[0];
        const float c = 0.8975979010256552f;  // 2*pi/7
        float sP = sob_b[0], sN = sob_b[0], s0 = sob_b[0];
#pragma unroll
        for (int h = 0; h < 32; ++h) {
            const float w2 = sob_w[h], a = soa_w[h], b = soa_b[h];
            sP += w2 * __sinf(c * (a * v + b));
            sN += w2 * __sinf(c * (-a * v + b));
            s0 += w2 * __sinf(c * b);
        }
        const float m  = fmaxf(s0, fmaxf(sP, sN));
        const float eP = __expf(sP - m), eN = __expf(sN - m), e0 = __expf(s0 - m);
        float r;
        if (sP > s0 && sN > s0)      r = 4.0f * v * (eP - eN) / (4.0f * eP + 4.0f * eN + e0);
        else if (sP > s0)            r = 4.0f * v * eP / (4.0f * eP + 5.0f * e0);
        else if (sN > s0)            r = -4.0f * v * eN / (4.0f * eN + 5.0f * e0);
        else                         r = 0.0f;
        out[n] = r;
    }
}

// ---- host-side orchestration ------------------------------------------------

extern "C" void kernel_launch(void* const* d_in, const int* in_sizes, int n_in,
                              void* d_out, int out_size, void* d_ws, size_t ws_size,
                              hipStream_t stream) {
    const float* x     = (const float*)d_in[0];   // [B,C,D] = [64,64,1024]
    const float* W1    = (const float*)d_in[1];   // [H,D]
    const float* W2    = (const float*)d_in[2];   // [H,H]
    const float* Wout  = (const float*)d_in[3];   // [1,H]
    const float* soa_w = (const float*)d_in[12];  // [32,1]
    const float* soa_b = (const float*)d_in[13];  // [32]
    const float* sob_w = (const float*)d_in[14];  // [1,32]
    const float* sob_b = (const float*)d_in[15];  // [1]

    const int H = 1024;
    const int D = in_sizes[1] / H;     // = 1024
    const int M = in_sizes[0] / D;     // = 4096 (B*C)

    char* ws = (char*)d_ws;
    _Float16* Xh  = (_Float16*)(ws + 0);                      // 8MB, dead after GEMM1
    _Float16* Th  = (_Float16*)(ws + (size_t)M * H * 2);      // 8MB
    _Float16* W1t = (_Float16*)(ws + (size_t)2 * M * H * 2);  // 2MB, [D][H]
    _Float16* W2t = W1t + (size_t)H * D;                      // 2MB, [H][H]
    _Float16* Uh  = Xh;                                       // alias Xh

    dim3 blk(256);
    cvt_f32_f16<<<dim3((M * D / 4 + 255) / 256), blk, 0, stream>>>(x, Xh, M * D / 4);
    cvt_transpose_f32_f16<<<dim3(D / 32, H / 32), blk, 0, stream>>>(W1, W1t, H, D);
    cvt_transpose_f32_f16<<<dim3(H / 32, H / 32), blk, 0, stream>>>(W2, W2t, H, H);

    dim3 g1(M / TILE_M, H / TILE_N);
    gemm_f16_tanh<<<g1, blk, 0, stream>>>(Xh, W1t, Th, M, H, D);
    gemm_f16_tanh<<<g1, blk, 0, stream>>>(Th, W2t, Uh, M, H, H);
    finalize_kernel<<<M, blk, 0, stream>>>(Uh, Wout, soa_w, soa_b, sob_w, sob_b,
                                           (float*)d_out, H);
}
